// MultiHeadSelfAttention_58540404244981
// MI455X (gfx1250) — matrix operand order repaired and checked by reading, not compiled, not run on hardware
//
#include <hip/hip_runtime.h>

typedef __bf16 bf16;
typedef __attribute__((ext_vector_type(16))) __bf16 v16bf;
typedef __attribute__((ext_vector_type(8)))  __bf16 v8bf;
typedef __attribute__((ext_vector_type(8)))  float  v8f;
typedef __attribute__((ext_vector_type(4)))  float  f32x4;
typedef __attribute__((ext_vector_type(4)))  int    i32x4;

typedef __attribute__((address_space(1))) i32x4 gi32x4;   // global 128-bit
typedef __attribute__((address_space(3))) i32x4 li32x4;   // LDS 128-bit

#define B_  2
#define S_  2048
#define D_  2048
#define H_  16
#define HD_ 128

#if __has_builtin(__builtin_amdgcn_global_load_async_to_lds_b128)
#define ASYNC_MODE 1
#else
#define ASYNC_MODE 0
#endif

// 16B global -> LDS copy; async (ASYNCcnt, no VGPR round-trip) when available
static __device__ __forceinline__ void async_copy16(const bf16* g, bf16* l) {
#if ASYNC_MODE
  __builtin_amdgcn_global_load_async_to_lds_b128((gi32x4*)g, (li32x4*)l, 0, 0);
#else
  *(v8bf*)l = *(const v8bf*)g;
#endif
}

static __device__ __forceinline__ void wait_async() {
#if ASYNC_MODE
#if __has_builtin(__builtin_amdgcn_s_wait_asynccnt)
  __builtin_amdgcn_s_wait_asynccnt(0);
#else
  asm volatile("s_wait_asynccnt 0x0" ::: "memory");
#endif
#endif
}

// round-to-nearest-even fp32 -> bf16 (integer path, toolchain-portable)
static __device__ __forceinline__ bf16 f2bf(float f) {
  unsigned u = __builtin_bit_cast(unsigned, f);
  u += 0x7FFFu + ((u >> 16) & 1u);
  unsigned short h = (unsigned short)(u >> 16);
  return __builtin_bit_cast(bf16, h);
}

// assemble a 16x32 (A) or 32x16 (B) bf16 WMMA fragment from two 16B chunks
static __device__ __forceinline__ v16bf ldfrag(const bf16* lo, const bf16* hi) {
  v8bf a = *(const v8bf*)lo;
  v8bf b = *(const v8bf*)hi;
  v16bf r;
#pragma unroll
  for (int i = 0; i < 8; ++i) { r[i] = a[i]; r[i + 8] = b[i]; }
  return r;
}

// ---------------------------------------------------------------------------
// One-time fp32 -> bf16 conversion (memory-bound; keeps GEMM loops VALU-free)
// ---------------------------------------------------------------------------
__global__ __launch_bounds__(256) void cvt_kernel(const float* __restrict__ s,
                                                  bf16* __restrict__ d) {
  const size_t i = ((size_t)blockIdx.x * 256 + threadIdx.x) * 8;
  f32x4 a = *(const f32x4*)(s + i);
  f32x4 b = *(const f32x4*)(s + i + 4);
  v8bf o;
#pragma unroll
  for (int e = 0; e < 4; ++e) { o[e] = f2bf(a[e]); o[e + 4] = f2bf(b[e]); }
  *(v8bf*)(d + i) = o;
}

// ---------------------------------------------------------------------------
// GEMM: out = A @ W^T  (A,W bf16; A: [4096 x 2048], W: [2048 x 2048])
//   mode 0/1: -> Q/K bf16 [B][H][S][hd], fused RoPE
//   mode 2:   -> V bf16 TRANSPOSED [B][H][hd][S]
//   mode 3:   -> fp32 [4096 x 2048] (final projection)
// Block 128x128, BK=32, 8 waves, wave tile 64x32. Double-buffered LDS with
// async global->LDS staging overlapped with WMMA on the other buffer.
// ---------------------------------------------------------------------------
__global__ __launch_bounds__(256) void fused_gemm_kernel(
    const bf16* __restrict__ A,
    const bf16* __restrict__ W0, const bf16* __restrict__ W1,
    const bf16* __restrict__ W2,
    bf16* __restrict__ OutQ, bf16* __restrict__ OutK, bf16* __restrict__ OutVt,
    float* __restrict__ OutF, int mode_base)
{
  __shared__ __attribute__((aligned(16))) union {
    struct { bf16 A[2][128][40]; bf16 B[2][128][40]; } t;  // 40960B ping-pong
    bf16  C[128][136];                                     // epilogue staging
    float Cf[64][136];
  } sh;

  const int mode = mode_base + (int)blockIdx.z;
  const bf16* __restrict__ W = (mode == 1) ? W1 : (mode == 2) ? W2 : W0;

  const int t    = threadIdx.x;
  const int lane = t & 31;
  const int wave = t >> 5;
  const int half = lane >> 4;
  const int l16  = lane & 15;
  const int wy   = wave >> 2;   // 0..1 (M)
  const int wx   = wave & 3;    // 0..3 (N)

  const int nbase   = (int)blockIdx.x * 128;   // == head*HD for modes 0..2
  const int rowbase = (int)blockIdx.y * 128;   // flat (b,s) row

  v8f acc[4][2] = {};

  const int lr = t >> 1;         // staging row 0..127
  const int lc = (t & 1) * 16;   // staging col chunk

  const bf16* __restrict__ Ap = A + (size_t)(rowbase + lr) * D_ + lc;
  const bf16* __restrict__ Wp = W + (size_t)(nbase + lr) * D_ + lc;

  auto issue = [&](int k0, int buf) {
    async_copy16(Ap + k0,     &sh.t.A[buf][lr][lc]);
    async_copy16(Ap + k0 + 8, &sh.t.A[buf][lr][lc + 8]);
    async_copy16(Wp + k0,     &sh.t.B[buf][lr][lc]);
    async_copy16(Wp + k0 + 8, &sh.t.B[buf][lr][lc + 8]);
  };

  issue(0, 0);
  for (int kt = 0; kt < D_ / 32; ++kt) {
    const int cur = kt & 1;
    wait_async();                 // my 4 outstanding copies for tile kt done
    __syncthreads();              // all waves' copies landed; prev buf free
    if (kt + 1 < D_ / 32) issue((kt + 1) * 32, cur ^ 1);

    v16bf af[4], bfr[2];
#pragma unroll
    for (int fm = 0; fm < 4; ++fm) {
      const int m = wy * 64 + fm * 16 + l16;
      af[fm] = ldfrag(&sh.t.A[cur][m][half * 8], &sh.t.A[cur][m][16 + half * 8]);
    }
#pragma unroll
    for (int fn = 0; fn < 2; ++fn) {
      const int n = wx * 32 + fn * 16 + l16;
      bfr[fn] = ldfrag(&sh.t.B[cur][n][half * 8], &sh.t.B[cur][n][16 + half * 8]);
    }
#pragma unroll
    for (int fm = 0; fm < 4; ++fm)
#pragma unroll
      for (int fn = 0; fn < 2; ++fn)
        acc[fm][fn] = __builtin_amdgcn_wmma_f32_16x16x32_bf16(
            false, af[fm], false, bfr[fn], (short)0, acc[fm][fn], false, false);
  }

  // ---------------- epilogue ----------------
  if (mode <= 1) {
    // Fused RoPE: positions = arange(S); tile N dim == head dim (BN==HD).
    // Pair partner column is one lane over -> single shfl_xor(1).
#pragma unroll
    for (int fm = 0; fm < 4; ++fm)
#pragma unroll
      for (int fn = 0; fn < 2; ++fn)
#pragma unroll
        for (int i = 0; i < 8; ++i) {
          float v   = acc[fm][fn][i];
          float prt = __shfl_xor(v, 1, 32);
          const int rl = wy * 64 + fm * 16 + i + 8 * half;
          const int d  = wx * 32 + fn * 16 + l16;       // 0..127 within head
          const int s  = (rowbase + rl) & (S_ - 1);
          const float invf = __expf((float)(d >> 1) * (-9.210340371976184f / 64.0f));
          const float ang  = (float)s * invf;
          const float cs = __cosf(ang), sn = __sinf(ang);
          acc[fm][fn][i] = (d & 1) ? (prt * sn + v * cs) : (v * cs - prt * sn);
        }
  }

  if (mode < 3) {
    __syncthreads();   // A/B buffers dead; C overlays them
#pragma unroll
    for (int fm = 0; fm < 4; ++fm)
#pragma unroll
      for (int fn = 0; fn < 2; ++fn)
#pragma unroll
        for (int i = 0; i < 8; ++i)
          sh.C[wy * 64 + fm * 16 + i + 8 * half][wx * 32 + fn * 16 + l16] =
              f2bf(acc[fm][fn][i]);
    __syncthreads();

    const int h = (int)blockIdx.x;
    if (mode <= 1) {
      bf16* __restrict__ dst = mode ? OutK : OutQ;
      const int sl = t >> 1, ch = (t & 1) * 64;
      const int rowg = rowbase + sl;
      const int bb = rowg >> 11, s = rowg & (S_ - 1);
      bf16* p = dst + (((size_t)bb * H_ + h) * S_ + s) * HD_ + ch;
#pragma unroll
      for (int j = 0; j < 8; ++j)
        *(v8bf*)(p + j * 8) = *(const v8bf*)&sh.C[sl][ch + j * 8];
    } else {
      // V stored transposed: [B][H][hd][S]  (column of C -> contiguous row)
      const int d = t >> 1, shalf = (t & 1) * 64;
      const int bb = rowbase >> 11, s0 = rowbase & (S_ - 1);
      bf16* p = OutVt + (((size_t)bb * H_ + h) * HD_ + d) * S_ + s0 + shalf;
#pragma unroll
      for (int j = 0; j < 8; ++j) {
        v8bf v;
#pragma unroll
        for (int e = 0; e < 8; ++e) v[e] = sh.C[shalf + j * 8 + e][d];
        *(v8bf*)(p + j * 8) = v;
      }
    }
  } else {
    // fp32 output: stage 64 rows at a time for vectorized b128 stores
    float* Csf = &sh.Cf[0][0];
#pragma unroll
    for (int pass = 0; pass < 2; ++pass) {
      __syncthreads();
      if (wy == pass) {
#pragma unroll
        for (int fm = 0; fm < 4; ++fm)
#pragma unroll
          for (int fn = 0; fn < 2; ++fn)
#pragma unroll
            for (int i = 0; i < 8; ++i)
              Csf[(fm * 16 + i + 8 * half) * 136 + wx * 32 + fn * 16 + l16] =
                  acc[fm][fn][i];
      }
      __syncthreads();
      const int r = t >> 2, q = (t & 3) * 32;
      float* p = OutF + (size_t)(rowbase + pass * 64 + r) * D_ + nbase + q;
      const float* srcf = Csf + r * 136 + q;
#pragma unroll
      for (int j = 0; j < 8; ++j)
        *(f32x4*)(p + j * 4) = *(const f32x4*)(srcf + j * 4);
    }
  }
}

// ---------------------------------------------------------------------------
// Flash attention: one block per (b, h, 128-query tile); 8 waves x 16 q-rows.
// Q fragments live in VGPRs for the whole kernel. Causal 64-wide K tiles,
// K/V staged via async global->LDS copies.
// ---------------------------------------------------------------------------
__global__ __launch_bounds__(256) void attn_kernel(
    const bf16* __restrict__ Qb, const bf16* __restrict__ Kb,
    const bf16* __restrict__ Vt, bf16* __restrict__ Ob)
{
  __shared__ __attribute__((aligned(16))) bf16 Ks[64][136];    // [k][d]
  __shared__ __attribute__((aligned(16))) bf16 Vs[128][72];    // [d][k]
  __shared__ __attribute__((aligned(16))) bf16 Ps[8][16][72];  // per-wave P

  const int t = threadIdx.x, lane = t & 31, w = t >> 5;
  const int half = lane >> 4, l16 = lane & 15;
  const int qt = blockIdx.x, h = blockIdx.y, b = blockIdx.z;
  const size_t bh = (size_t)b * H_ + h;
  const bf16* __restrict__ Qp = Qb + bh * (size_t)S_ * HD_;
  const bf16* __restrict__ Kp = Kb + bh * (size_t)S_ * HD_;
  const bf16* __restrict__ Vp = Vt + bh * (size_t)HD_ * S_;

  const int qr0 = qt * 128 + w * 16;

  v16bf aq[4];
  {
    const bf16* qrow = Qp + (size_t)(qr0 + l16) * HD_;
#pragma unroll
    for (int dc = 0; dc < 4; ++dc)
      aq[dc] = ldfrag(qrow + dc * 32 + half * 8, qrow + dc * 32 + 16 + half * 8);
  }

  v8f o[8] = {};
  float mrun[8], lrun[8];
#pragma unroll
  for (int i = 0; i < 8; ++i) { mrun[i] = -1e30f; lrun[i] = 0.f; }

  const int rK = t >> 2, cK = (t & 3) * 32;   // K-stage coords
  const int dV = t >> 1, cV = (t & 1) * 32;   // V-stage coords

  const int nkt = (qt + 1) * 2;
  for (int kt = 0; kt < nkt; ++kt) {
    const int k0 = kt * 64;
    __syncthreads();   // everyone done with previous Ks/Vs
    {
      const bf16* src = Kp + (size_t)(k0 + rK) * HD_ + cK;
#pragma unroll
      for (int j = 0; j < 4; ++j) async_copy16(src + j * 8, &Ks[rK][cK + j * 8]);
    }
    {
      const bf16* src = Vp + (size_t)dV * S_ + k0 + cV;
#pragma unroll
      for (int j = 0; j < 4; ++j) async_copy16(src + j * 8, &Vs[dV][cV + j * 8]);
    }
    wait_async();
    __syncthreads();

    // S = Q @ K^T
    v8f sc[4] = {};
#pragma unroll
    for (int dc = 0; dc < 4; ++dc)
#pragma unroll
      for (int kn = 0; kn < 4; ++kn) {
        v16bf bk = ldfrag(&Ks[kn * 16 + l16][dc * 32 + half * 8],
                          &Ks[kn * 16 + l16][dc * 32 + 16 + half * 8]);
        sc[kn] = __builtin_amdgcn_wmma_f32_16x16x32_bf16(
            false, aq[dc], false, bk, (short)0, sc[kn], false, false);
      }

    // online softmax
    const float SCALE = 0.08838834764831845f;  // 1/sqrt(128)
    float px[4][8];
#pragma unroll
    for (int i = 0; i < 8; ++i) {
      const int qr = qr0 + i + 8 * half;
      float m = -1e30f;
#pragma unroll
      for (int kn = 0; kn < 4; ++kn) {
        float v = sc[kn][i] * SCALE;
        const int kc = k0 + kn * 16 + l16;
        if (kc > qr) v = -1e30f;               // causal mask
        px[kn][i] = v;
        m = fmaxf(m, v);
      }
#pragma unroll
      for (int off = 1; off < 16; off <<= 1) m = fmaxf(m, __shfl_xor(m, off, 32));
      const float nm = fmaxf(mrun[i], m);
      const float alpha = __expf(mrun[i] - nm);
      mrun[i] = nm;
      float rs = 0.f;
#pragma unroll
      for (int kn = 0; kn < 4; ++kn) {
        const float p = __expf(px[kn][i] - nm);
        px[kn][i] = p;
        rs += p;
      }
#pragma unroll
      for (int off = 1; off < 16; off <<= 1) rs += __shfl_xor(rs, off, 32);
      lrun[i] = lrun[i] * alpha + rs;
#pragma unroll
      for (int od = 0; od < 8; ++od) o[od][i] *= alpha;
    }

    // restage P (C layout -> A layout) through per-wave LDS region
#pragma unroll
    for (int kn = 0; kn < 4; ++kn)
#pragma unroll
      for (int i = 0; i < 8; ++i)
        Ps[w][i + 8 * half][kn * 16 + l16] = f2bf(px[kn][i]);

    // O += P @ V
#pragma unroll
    for (int kc = 0; kc < 2; ++kc) {
      v16bf ap = ldfrag(&Ps[w][l16][kc * 32 + half * 8],
                        &Ps[w][l16][kc * 32 + 16 + half * 8]);
#pragma unroll
      for (int od = 0; od < 8; ++od) {
        v16bf bv = ldfrag(&Vs[od * 16 + l16][kc * 32 + half * 8],
                          &Vs[od * 16 + l16][kc * 32 + 16 + half * 8]);
        o[od] = __builtin_amdgcn_wmma_f32_16x16x32_bf16(
            false, ap, false, bv, (short)0, o[od], false, false);
      }
    }
  }

  // normalize and merge heads -> [B][S][D] bf16
  float inv[8];
#pragma unroll
  for (int i = 0; i < 8; ++i) inv[i] = 1.f / lrun[i];
#pragma unroll
  for (int od = 0; od < 8; ++od)
#pragma unroll
    for (int i = 0; i < 8; ++i) {
      const int qr = qr0 + i + 8 * half;
      const int d = od * 16 + l16;
      Ob[((size_t)b * S_ + qr) * D_ + h * HD_ + d] = f2bf(o[od][i] * inv[i]);
    }
}

extern "C" void kernel_launch(void* const* d_in, const int* in_sizes, int n_in,
                              void* d_out, int out_size, void* d_ws, size_t ws_size,
                              hipStream_t stream) {
  (void)in_sizes; (void)n_in; (void)out_size; (void)ws_size;
  const float* x  = (const float*)d_in[0];
  // d_in[1] = token_positions: unused by the reference (it uses arange(S))
  const float* Wq = (const float*)d_in[2];
  const float* Wk = (const float*)d_in[3];
  const float* Wv = (const float*)d_in[4];
  const float* Wo = (const float*)d_in[5];
  float* out = (float*)d_out;

  const size_t n = (size_t)B_ * S_ * D_;   // 8,388,608
  const size_t m = (size_t)D_ * D_;        // 4,194,304
  bf16* Qb  = (bf16*)d_ws;
  bf16* Kb  = Qb + n;
  bf16* Vtb = Kb + n;
  bf16* Ob  = Vtb + n;
  bf16* Xb  = Ob + n;
  bf16* Wqb = Xb + n;
  bf16* Wkb = Wqb + m;
  bf16* Wvb = Wkb + m;
  bf16* Wob = Wvb + m;                     // total ~117 MB of workspace

  dim3 blk(256);
  // one-time fp32 -> bf16 down-conversion (memory-bound)
  cvt_kernel<<<dim3((unsigned)(n / 2048)), blk, 0, stream>>>(x,  Xb);
  cvt_kernel<<<dim3((unsigned)(m / 2048)), blk, 0, stream>>>(Wq, Wqb);
  cvt_kernel<<<dim3((unsigned)(m / 2048)), blk, 0, stream>>>(Wk, Wkb);
  cvt_kernel<<<dim3((unsigned)(m / 2048)), blk, 0, stream>>>(Wv, Wvb);
  cvt_kernel<<<dim3((unsigned)(m / 2048)), blk, 0, stream>>>(Wo, Wob);
  // QKV projections with fused RoPE (z selects Q/K/V)
  fused_gemm_kernel<<<dim3(16, 32, 3), blk, 0, stream>>>(
      Xb, Wqb, Wkb, Wvb, Qb, Kb, Vtb, nullptr, 0);
  // flash attention
  attn_kernel<<<dim3(16, 16, 2), blk, 0, stream>>>(Qb, Kb, Vtb, Ob);
  // output projection -> fp32
  fused_gemm_kernel<<<dim3(16, 32, 1), blk, 0, stream>>>(
      Ob, Wob, Wob, Wob, nullptr, nullptr, nullptr, out, 3);
}
